// CrossAttentionBlock_75857712382538
// MI455X (gfx1250) — matrix operand order repaired and checked by reading, not compiled, not run on hardware
//
#include <hip/hip_runtime.h>
#include <hip/hip_bf16.h>

// ---------------------------------------------------------------------------
// CrossAttentionBlock for MI455X (gfx1250): bf16 WMMA everywhere, f32 accum.
// Async global->LDS staging (ASYNCcnt) for GEMM panels and attention K tiles.
// ---------------------------------------------------------------------------

typedef __attribute__((ext_vector_type(16))) __bf16 v16bf;
typedef __attribute__((ext_vector_type(8)))  float  v8f;

#define B_   2
#define SEQ  2048
#define CH   1024
#define NH   16
#define DH   64
#define BN   (B_ * SEQ)   // 4096 rows total

__device__ __forceinline__ unsigned short f2bf(float f) {
    union { float f; unsigned int u; } c; c.f = f;
    unsigned int u = c.u;
    u += 0x7fffu + ((u >> 16) & 1u);   // round-to-nearest-even
    return (unsigned short)(u >> 16);
}

// LDS byte offset of a generic pointer to a __shared__ object
__device__ __forceinline__ unsigned lds_addr(const void* p) {
    return (unsigned)(unsigned long long)
        (__attribute__((address_space(3))) const char*)p;
}

union FragAB { v16bf v; uint4 q[2]; unsigned short u[16]; };

// A operand (16x32, bf16): lane&15 = row M; kb=(lane>>4)*8; K = {kb..kb+7, kb+16..kb+23}
__device__ __forceinline__ void loadA(const unsigned short* base, int stride, FragAB& f) {
    const int l = threadIdx.x & 31;
    const unsigned short* p = base + (l & 15) * stride + ((l >> 4) << 3);
    f.q[0] = *(const uint4*)(p);
    f.q[1] = *(const uint4*)(p + 16);
}

// B operand (32x16, bf16): lane&15 = col N; kb=(lane>>4)*8; K = {kb..kb+7, kb+16..kb+23}
__device__ __forceinline__ void loadB(const unsigned short* base, int stride, FragAB& f) {
    const int l = threadIdx.x & 31;
    const unsigned short* p = base + (l & 15) * stride + ((l >> 4) << 3);
    f.q[0] = *(const uint4*)(p);
    f.q[1] = *(const uint4*)(p + 16);
}

__device__ __forceinline__ v8f wmma_bf16(const FragAB& a, const FragAB& b, v8f c) {
    return __builtin_amdgcn_wmma_f32_16x16x32_bf16(false, a.v, false, b.v,
                                                   (short)0, c, false, false);
}

// ---------------------------------------------------------------------------
// K0: f32 -> bf16 conversion (vectorized)
// ---------------------------------------------------------------------------
__global__ void __launch_bounds__(256)
cvt_bf16(const float4* __restrict__ in, uint2* __restrict__ out, int n4) {
    int i = blockIdx.x * 256 + threadIdx.x;
    if (i < n4) {
        float4 v = in[i];
        uint2 o;
        o.x = (unsigned int)f2bf(v.x) | ((unsigned int)f2bf(v.y) << 16);
        o.y = (unsigned int)f2bf(v.z) | ((unsigned int)f2bf(v.w) << 16);
        out[i] = o;
    }
}

// ---------------------------------------------------------------------------
// K1/K2/K5: C(f32)[M,N] = A(bf16)[M,K] @ W(bf16)[N,K]^T + bias
// 128-thread blocks (4 waves), block tile 128x128, wave tile 64x64.
// K staged 32-wide through LDS with global_load_async_to_lds_b128.
// Per k-step per wave: 8 fragment loads -> 16 WMMAs.
// ---------------------------------------------------------------------------
__global__ void __launch_bounds__(128)
gemm_bf16(const unsigned short* __restrict__ A, const unsigned short* __restrict__ W,
          const float* __restrict__ bias, float* __restrict__ Cout, int K, int N) {
    __shared__ unsigned short sA[128 * 40];
    __shared__ unsigned short sB[128 * 40];

    const int t = threadIdx.x;
    const int w = t >> 5, lane = t & 31;
    const int wm = w & 1;    // 2 wave-rows of 64
    const int wn = w >> 1;   // 2 wave-cols of 64
    const int row0 = blockIdx.y * 128;
    const int col0 = blockIdx.x * 128;

    v8f acc[4][4];
    #pragma unroll
    for (int nt = 0; nt < 4; ++nt) {
        float bv = bias[col0 + wn * 64 + nt * 16 + (lane & 15)];
        #pragma unroll
        for (int mt = 0; mt < 4; ++mt)
            #pragma unroll
            for (int j = 0; j < 8; ++j) acc[mt][nt][j] = bv;
    }

    // each thread stages one full 32-elem (64B) row of A and of B per k-step
    const unsigned short* Ag = A + (size_t)(row0 + t) * K;
    const unsigned short* Wg = W + (size_t)(col0 + t) * K;
    const unsigned ldsA = lds_addr(&sA[t * 40]);
    const unsigned ldsB = lds_addr(&sB[t * 40]);

    for (int k0 = 0; k0 < K; k0 += 32) {
        __syncthreads();
        asm volatile(
            "global_load_async_to_lds_b128 %0, %2, off\n\t"
            "global_load_async_to_lds_b128 %0, %2, off offset:16\n\t"
            "global_load_async_to_lds_b128 %0, %2, off offset:32\n\t"
            "global_load_async_to_lds_b128 %0, %2, off offset:48\n\t"
            "global_load_async_to_lds_b128 %1, %3, off\n\t"
            "global_load_async_to_lds_b128 %1, %3, off offset:16\n\t"
            "global_load_async_to_lds_b128 %1, %3, off offset:32\n\t"
            "global_load_async_to_lds_b128 %1, %3, off offset:48"
            :: "v"(ldsA), "v"(ldsB), "v"(Ag + k0), "v"(Wg + k0)
            : "memory");
        asm volatile("s_wait_asynccnt 0" ::: "memory");
        __syncthreads();

        FragAB af[4];
        #pragma unroll
        for (int mt = 0; mt < 4; ++mt)
            loadA(&sA[(wm * 64 + mt * 16) * 40], 40, af[mt]);
        #pragma unroll
        for (int nt = 0; nt < 4; ++nt) {
            FragAB bfr;
            loadB(&sB[(wn * 64 + nt * 16) * 40], 40, bfr);
            #pragma unroll
            for (int mt = 0; mt < 4; ++mt)
                acc[mt][nt] = wmma_bf16(af[mt], bfr, acc[mt][nt]);
        }
    }

    #pragma unroll
    for (int mt = 0; mt < 4; ++mt)
        #pragma unroll
        for (int nt = 0; nt < 4; ++nt)
            #pragma unroll
            for (int j = 0; j < 8; ++j) {
                int r = row0 + wm * 64 + mt * 16 + j + ((lane >> 4) << 3);
                int c = col0 + wn * 64 + nt * 16 + (lane & 15);
                Cout[(size_t)r * N + c] = acc[mt][nt][j];
            }
}

// ---------------------------------------------------------------------------
// K3: per-head LayerNorm + pack to bf16 head-major layouts.
// One wave per (b,h,row) group of 64; kinds: 0=q(LN*scale) 1=k(LN) 2=v(copy)
// ---------------------------------------------------------------------------
__global__ void __launch_bounds__(256)
ln_pack(const float* __restrict__ qf, const float* __restrict__ kvf,
        const float* __restrict__ qg, const float* __restrict__ qbeta,
        const float* __restrict__ kg, const float* __restrict__ kbeta,
        unsigned short* __restrict__ qo, unsigned short* __restrict__ ko,
        unsigned short* __restrict__ vo) {
    int gw   = blockIdx.x * 8 + (threadIdx.x >> 5);
    int lane = threadIdx.x & 31;
    int kind = gw >> 16;        // 65536 head-rows per kind
    int idx  = gw & 65535;
    int b = idx >> 15;
    int h = (idx >> 11) & 15;
    int n = idx & 2047;

    const float* src;
    const float *g = nullptr, *be = nullptr;
    float scale = 1.0f;
    if (kind == 0)      { src = qf  + ((size_t)(b * SEQ + n)) * CH        + h * DH; g = qg; be = qbeta; scale = 0.125f; }
    else if (kind == 1) { src = kvf + ((size_t)(b * SEQ + n)) * (2 * CH)  + h * DH; g = kg; be = kbeta; }
    else                { src = kvf + ((size_t)(b * SEQ + n)) * (2 * CH) + CH + h * DH; }

    float2 xv = *(const float2*)(src + 2 * lane);
    float a0 = xv.x, a1 = xv.y;
    if (kind < 2) {
        float s = a0 + a1, s2 = a0 * a0 + a1 * a1;
        for (int off = 16; off; off >>= 1) {
            s  += __shfl_xor(s,  off);
            s2 += __shfl_xor(s2, off);
        }
        float mu   = s * (1.0f / 64.0f);
        float var  = s2 * (1.0f / 64.0f) - mu * mu;
        float rstd = rsqrtf(var + 1e-5f);
        float g0 = g[2 * lane], g1 = g[2 * lane + 1];
        float b0 = be[2 * lane], b1 = be[2 * lane + 1];
        a0 = ((a0 - mu) * rstd * g0 + b0) * scale;
        a1 = ((a1 - mu) * rstd * g1 + b1) * scale;
    }
    unsigned short* dst = ((kind == 0) ? qo : (kind == 1) ? ko : vo)
                        + ((size_t)((b * NH + h) * SEQ + n)) * DH + 2 * lane;
    *(unsigned int*)dst = (unsigned int)f2bf(a0) | ((unsigned int)f2bf(a1) << 16);
}

// ---------------------------------------------------------------------------
// K4: flash attention. Block = (b,h, 128 q rows); wave owns 16 q rows.
// K chunk staged async to LDS; V chunk register-transposed into LDS.
// Online softmax in f32, P restaged through per-wave LDS slab for WMMA A-layout.
// ---------------------------------------------------------------------------
__global__ void __launch_bounds__(256)
attn_kernel(const unsigned short* __restrict__ Q, const unsigned short* __restrict__ Kb,
            const unsigned short* __restrict__ Vb, unsigned short* __restrict__ Ob) {
    __shared__ unsigned short sK[64 * 72];       // [key][dim]
    __shared__ unsigned short sV[64 * 72];       // transposed: [dim][key]
    __shared__ unsigned short sP[8][16 * 72];    // per-wave P staging

    const int t = threadIdx.x, w = t >> 5, lane = t & 31;
    const int bh = blockIdx.x >> 4;              // b*16 + h
    const int n0 = (blockIdx.x & 15) * 128 + w * 16;

    // Q fragments live in registers for the whole kernel
    const unsigned short* qbase = Q + ((size_t)bh * SEQ + n0) * DH;
    FragAB qf0, qf1;
    loadA(qbase +  0, DH, qf0);
    loadA(qbase + 32, DH, qf1);

    v8f Oacc[4];
    float mrow[8], lrow[8];
    #pragma unroll
    for (int dt = 0; dt < 4; ++dt)
        #pragma unroll
        for (int j = 0; j < 8; ++j) Oacc[dt][j] = 0.f;
    #pragma unroll
    for (int j = 0; j < 8; ++j) { mrow[j] = -1e30f; lrow[j] = 0.f; }

    const unsigned short* kgbase = Kb + (size_t)bh * SEQ * DH;
    const unsigned short* vgbase = Vb + (size_t)bh * SEQ * DH;
    const int lm = t >> 2;          // 0..63 key row in chunk
    const int ld = (t & 3) << 4;    // 0,16,32,48 dim base
    const unsigned ldsK = lds_addr(&sK[lm * 72 + ld]);

    for (int m0 = 0; m0 < SEQ; m0 += 64) {
        __syncthreads();
        {   // stage K chunk row-major: async global->LDS (32B per thread)
            const unsigned short* src = kgbase + (size_t)(m0 + lm) * DH + ld;
            asm volatile(
                "global_load_async_to_lds_b128 %0, %1, off\n\t"
                "global_load_async_to_lds_b128 %0, %1, off offset:16"
                :: "v"(ldsK), "v"(src) : "memory");
            // stage V chunk transposed [dim][key] (needs VGPR round trip)
            const unsigned short* vs = vgbase + (size_t)(m0 + lm) * DH + ld;
            union { uint4 q[2]; unsigned short u[16]; } vv;
            vv.q[0] = *(const uint4*)(vs);
            vv.q[1] = *(const uint4*)(vs + 8);
            #pragma unroll
            for (int i = 0; i < 16; ++i) sV[(ld + i) * 72 + lm] = vv.u[i];
        }
        asm volatile("s_wait_asynccnt 0" ::: "memory");
        __syncthreads();

        // S = Q @ K^T over 4 key sub-tiles of 16
        v8f S[4];
        #pragma unroll
        for (int nt = 0; nt < 4; ++nt) {
            FragAB b0, b1;
            loadB(&sK[(nt * 16) * 72 +  0], 72, b0);
            loadB(&sK[(nt * 16) * 72 + 32], 72, b1);
            v8f z;
            #pragma unroll
            for (int j = 0; j < 8; ++j) z[j] = 0.f;
            z = wmma_bf16(qf0, b0, z);
            z = wmma_bf16(qf1, b1, z);
            S[nt] = z;
        }

        // online softmax (rows live in 16-lane groups)
        float mx[8], fac[8], psum[8];
        #pragma unroll
        for (int j = 0; j < 8; ++j)
            mx[j] = fmaxf(fmaxf(S[0][j], S[1][j]), fmaxf(S[2][j], S[3][j]));
        for (int off = 1; off < 16; off <<= 1)
            #pragma unroll
            for (int j = 0; j < 8; ++j) mx[j] = fmaxf(mx[j], __shfl_xor(mx[j], off));
        #pragma unroll
        for (int j = 0; j < 8; ++j) {
            float mn = fmaxf(mrow[j], mx[j]);
            fac[j] = __expf(mrow[j] - mn);
            mrow[j] = mn;
            psum[j] = 0.f;
        }

        unsigned short* pw = &sP[w][0];
        #pragma unroll
        for (int nt = 0; nt < 4; ++nt)
            #pragma unroll
            for (int j = 0; j < 8; ++j) {
                float pv = __expf(S[nt][j] - mrow[j]);
                psum[j] += pv;
                pw[(j + ((lane >> 4) << 3)) * 72 + nt * 16 + (lane & 15)] = f2bf(pv);
            }
        for (int off = 1; off < 16; off <<= 1)
            #pragma unroll
            for (int j = 0; j < 8; ++j) psum[j] += __shfl_xor(psum[j], off);
        #pragma unroll
        for (int j = 0; j < 8; ++j) lrow[j] = lrow[j] * fac[j] + psum[j];
        #pragma unroll
        for (int dt = 0; dt < 4; ++dt)
            #pragma unroll
            for (int j = 0; j < 8; ++j) Oacc[dt][j] *= fac[j];

        // O += P @ V
        FragAB pf0, pf1;
        loadA(pw +  0, 72, pf0);
        loadA(pw + 32, 72, pf1);
        #pragma unroll
        for (int dt = 0; dt < 4; ++dt) {
            FragAB vf0, vf1;
            loadB(&sV[(dt * 16) * 72 +  0], 72, vf0);
            loadB(&sV[(dt * 16) * 72 + 32], 72, vf1);
            Oacc[dt] = wmma_bf16(pf0, vf0, Oacc[dt]);
            Oacc[dt] = wmma_bf16(pf1, vf1, Oacc[dt]);
        }
    }

    // normalize, store bf16 into [B,N,C] with head offset
    const int b = bh >> 4, h = bh & 15;
    float rinv[8];
    #pragma unroll
    for (int j = 0; j < 8; ++j) rinv[j] = 1.0f / lrow[j];
    #pragma unroll
    for (int dt = 0; dt < 4; ++dt)
        #pragma unroll
        for (int j = 0; j < 8; ++j) {
            int n = n0 + j + ((lane >> 4) << 3);
            int d = h * DH + dt * 16 + (lane & 15);
            Ob[((size_t)(b * SEQ + n)) * CH + d] = f2bf(Oacc[dt][j] * rinv[j]);
        }
}

// ---------------------------------------------------------------------------
extern "C" void kernel_launch(void* const* d_in, const int* in_sizes, int n_in,
                              void* d_out, int out_size, void* d_ws, size_t ws_size,
                              hipStream_t stream) {
    (void)in_sizes; (void)n_in; (void)out_size; (void)ws_size;
    const float* x     = (const float*)d_in[0];
    const float* y     = (const float*)d_in[1];
    const float* Wq    = (const float*)d_in[2];
    const float* bq    = (const float*)d_in[3];
    const float* Wkv   = (const float*)d_in[4];
    const float* bkv   = (const float*)d_in[5];
    const float* qg    = (const float*)d_in[6];
    const float* qbeta = (const float*)d_in[7];
    const float* kg    = (const float*)d_in[8];
    const float* kbeta = (const float*)d_in[9];
    const float* Wproj = (const float*)d_in[10];
    const float* bproj = (const float*)d_in[11];

    char* p = (char*)d_ws;
    auto alloc = [&](size_t bytes) -> char* {
        char* r = p; p += (bytes + 255) & ~(size_t)255; return r;
    };
    unsigned short* xb  = (unsigned short*)alloc((size_t)BN * CH * 2);
    unsigned short* yb  = (unsigned short*)alloc((size_t)BN * CH * 2);
    unsigned short* Wqb = (unsigned short*)alloc((size_t)CH * CH * 2);
    unsigned short* Wkb = (unsigned short*)alloc((size_t)2 * CH * CH * 2);
    unsigned short* Wpb = (unsigned short*)alloc((size_t)CH * CH * 2);
    unsigned short* qpk = (unsigned short*)alloc((size_t)BN * CH * 2);
    unsigned short* kpk = (unsigned short*)alloc((size_t)BN * CH * 2);
    unsigned short* vpk = (unsigned short*)alloc((size_t)BN * CH * 2);
    float* qf32  = (float*)alloc((size_t)BN * CH * 4);
    float* kvf32 = (float*)alloc((size_t)BN * 2 * CH * 4);
    // attention output (bf16) aliases qf32, which is dead after ln_pack
    unsigned short* aob = (unsigned short*)qf32;

    // K0: convert to bf16
    auto cvt = [&](const float* in, unsigned short* out, size_t count) {
        int n4 = (int)(count / 4);
        cvt_bf16<<<n4 / 256, 256, 0, stream>>>((const float4*)in, (uint2*)out, n4);
    };
    cvt(x,     xb,  (size_t)BN * CH);
    cvt(y,     yb,  (size_t)BN * CH);
    cvt(Wq,    Wqb, (size_t)CH * CH);
    cvt(Wkv,   Wkb, (size_t)2 * CH * CH);
    cvt(Wproj, Wpb, (size_t)CH * CH);

    // K1: q = x @ Wq^T + bq  (4096 x 1024)
    gemm_bf16<<<dim3(CH / 128, BN / 128), 128, 0, stream>>>(xb, Wqb, bq, qf32, CH, CH);
    // K2: kv = y @ Wkv^T + bkv (4096 x 2048)
    gemm_bf16<<<dim3(2 * CH / 128, BN / 128), 128, 0, stream>>>(yb, Wkb, bkv, kvf32, CH, 2 * CH);
    // K3: LayerNorm + pack (3 * 65536 head-rows, 8 waves/block)
    ln_pack<<<(3 * 65536) / 8, 256, 0, stream>>>(qf32, kvf32, qg, qbeta, kg, kbeta,
                                                 qpk, kpk, vpk);
    // K4: flash attention (B*H*16 n-tiles = 512 blocks)
    attn_kernel<<<B_ * NH * (SEQ / 128), 256, 0, stream>>>(qpk, kpk, vpk, aob);
    // K5: out = attn_out @ Wproj^T + bproj -> d_out (f32)
    gemm_bf16<<<dim3(CH / 128, BN / 128), 128, 0, stream>>>(aob, Wpb, bproj,
                                                            (float*)d_out, CH, CH);
}
